// IsolatedAttention_56195352101337
// MI455X (gfx1250) — compile-verified, matrix operand order checked by reading, not run on hardware
//
#include <hip/hip_runtime.h>

#define D_MODEL 1024
#define SEQ     4096
#define BATCH   4
#define LDSW    40   // padded LDS row stride in halfs (80B): conflict-free ds_load_b128

typedef __attribute__((ext_vector_type(16))) _Float16 v16h;
typedef __attribute__((ext_vector_type(8)))  _Float16 v8h;
typedef __attribute__((ext_vector_type(8)))  float    v8f;
typedef __attribute__((ext_vector_type(4)))  int      v4i;

#if __has_builtin(__builtin_amdgcn_global_load_async_to_lds_b128) && \
    __has_builtin(__builtin_amdgcn_s_wait_asynccnt)
#define USE_ASYNC_LDS 1
typedef __attribute__((address_space(1))) v4i* gv4i_p;   // global (device) int4*
typedef __attribute__((address_space(3))) v4i* lv4i_p;   // LDS (shared) int4*
#else
#define USE_ASYNC_LDS 0
#endif

#if __has_builtin(__builtin_amdgcn_sched_barrier)
#define SCHED_FENCE() __builtin_amdgcn_sched_barrier(0)
#else
#define SCHED_FENCE()
#endif

// ---------------------------------------------------------------------------
// Fragment gather from an LDS-staged 128x32 f16 tile (row stride LDSW halfs).
// CDNA5 16-bit A 16x32 / B 32x16 layout: lanes 0-15 hold row (lane&15) with
// K = {0..7, 16..23}; lanes 16-31 hold the same rows with K = {8..15, 24..31}.
// ---------------------------------------------------------------------------
__device__ __forceinline__ v16h frag_lds(const _Float16* tile, int row0) {
  const int lane = threadIdx.x & 31;
  const _Float16* base = tile + (row0 + (lane & 15)) * LDSW + ((lane >> 4) << 3);
  v8h lo = *(const v8h*)(base);        // ds_load_b128
  v8h hi = *(const v8h*)(base + 16);   // ds_load_b128
  return __builtin_shufflevector(lo, hi, 0, 1, 2, 3, 4, 5, 6, 7,
                                         8, 9, 10, 11, 12, 13, 14, 15);
}

__device__ __forceinline__ v8f wmma16(v16h a, v16h b, v8f c) {
  return __builtin_amdgcn_wmma_f32_16x16x32_f16(false, a, false, b,
                                                (short)0, c, false, false);
}

__device__ __forceinline__ v8f zero8() {
  v8f z = {0.0f, 0.0f, 0.0f, 0.0f, 0.0f, 0.0f, 0.0f, 0.0f};
  return z;
}

// Issue one 16B global->LDS copy (async on gfx1250, sync fallback otherwise).
__device__ __forceinline__ void copy16(const _Float16* gp, _Float16* lp) {
#if USE_ASYNC_LDS
  __builtin_amdgcn_global_load_async_to_lds_b128((gv4i_p)gp, (lv4i_p)lp, 0, 0);
#else
  *(v8h*)lp = *(const v8h*)gp;
#endif
}

// ---------------------------------------------------------------------------
// Unified f16 GEMM:  C[m,n] = sum_k A[m,k] * B[n,k]
// A: [gridDim.x*128][K], B: [gridDim.y*128][K].
// Block = 256 threads (8 waves); WG tile 128x128; double-buffered async-LDS
// pipeline; 8 accumulator chains per wave.  All 18 ds_load_b128 fragment
// reads are issued ahead of a sched_barrier so the 8 WMMAs run back-to-back
// with graduated dscnt waits.  __launch_bounds__(256,1) lifts the VGPR cap
// (~160 live regs) so the fragment set is NOT spilled to scratch; occupancy
// is LDS-bound (40KB/WG) anyway.
// Epilogue MODE: 0 = f32 store (ldC stride)
//                1 = f16 store, scaled by 1/sqrt(D)   (Q projection)
//                2 = f16 store                         (K projection)
//                3 = f16 store transposed per batch    (V -> Vt[b][col][s])
// ---------------------------------------------------------------------------
template <int MODE>
__global__ void __launch_bounds__(256, 1)
gemm_nt(const _Float16* __restrict__ A,
        const _Float16* __restrict__ B,
        void* __restrict__ Cv,
        int K, int ldC) {
  __shared__ _Float16 sA[2][128 * LDSW];
  __shared__ _Float16 sB[2][128 * LDSW];

  const int wave = threadIdx.x >> 5;
  const int lane = threadIdx.x & 31;
  const int mWG = blockIdx.x * 128;
  const int nWG = blockIdx.y * 128;

  // Per-thread staging assignment: 2 chunks of 16B per tile (512 x 16B total).
  const int c0   = threadIdx.x * 2;
  const int row0 = c0 >> 2,       kc0 = (c0 & 3) << 3;
  const int row1 = (c0 + 1) >> 2, kc1 = ((c0 + 1) & 3) << 3;

  const _Float16* gA0 = A + (size_t)(mWG + row0) * K + kc0;
  const _Float16* gA1 = A + (size_t)(mWG + row1) * K + kc1;
  const _Float16* gB0 = B + (size_t)(nWG + row0) * K + kc0;
  const _Float16* gB1 = B + (size_t)(nWG + row1) * K + kc1;
  const int lo0 = row0 * LDSW + kc0;
  const int lo1 = row1 * LDSW + kc1;

  v8f acc[8];
#pragma unroll
  for (int n = 0; n < 8; ++n) acc[n] = zero8();

  const int nsteps = K >> 5;   // K / 32

  // Prologue: stage k-step 0 into buffer 0.
  copy16(gA0, &sA[0][lo0]);
  copy16(gA1, &sA[0][lo1]);
  copy16(gB0, &sB[0][lo0]);
  copy16(gB1, &sB[0][lo1]);

  for (int s = 0; s < nsteps; ++s) {
    const int cur = s & 1;
    const int nxt = cur ^ 1;

#if USE_ASYNC_LDS
    // Stage-s copies were issued a full compute phase ago: free in steady state.
    __builtin_amdgcn_s_wait_asynccnt(0);
#endif
    __syncthreads();   // current-stage tiles visible to all waves

    // Issue next stage's copies; they overlap with this step's compute.
    // Buffer nxt was last read in compute step s-1, which is separated from
    // here by two barriers -> safe to overwrite.
    if (s + 1 < nsteps) {
      const int ko = (s + 1) << 5;   // k offset in halfs
      copy16(gA0 + ko, &sA[nxt][lo0]);
      copy16(gA1 + ko, &sA[nxt][lo1]);
      copy16(gB0 + ko, &sB[nxt][lo0]);
      copy16(gB1 + ko, &sB[nxt][lo1]);
    }

    // Issue ALL fragment loads first; the sched_barrier pins them above the
    // WMMAs, forcing distinct registers (no WAR recycling) and letting the
    // compiler use graduated s_wait_dscnt values.
    v16h a = frag_lds(&sA[cur][0], wave * 16);
    v16h bf[8];
#pragma unroll
    for (int n = 0; n < 8; ++n) bf[n] = frag_lds(&sB[cur][0], n * 16);
    SCHED_FENCE();
#pragma unroll
    for (int n = 0; n < 8; ++n) acc[n] = wmma16(a, bf[n], acc[n]);

    __syncthreads();   // all waves done reading before buffers are reused
  }

  const int rOff = (lane >> 4) << 3;   // 0 or 8
  const int cOff = lane & 15;
#pragma unroll
  for (int n = 0; n < 8; ++n) {
    const int col = nWG + n * 16 + cOff;
#pragma unroll
    for (int r = 0; r < 8; ++r) {
      const int row = mWG + wave * 16 + rOff + r;
      const float v = acc[n][r];
      if (MODE == 0) {
        ((float*)Cv)[(size_t)row * ldC + col] = v;
      } else if (MODE == 1) {
        ((_Float16*)Cv)[(size_t)row * ldC + col] = (_Float16)(v * 0.03125f);
      } else if (MODE == 2) {
        ((_Float16*)Cv)[(size_t)row * ldC + col] = (_Float16)v;
      } else {
        const int b = row >> 12;          // row / SEQ
        const int sIdx = row & (SEQ - 1);
        ((_Float16*)Cv)[((size_t)b * D_MODEL + col) * SEQ + sIdx] = (_Float16)v;
      }
    }
  }
}

// ---------------------------------------------------------------------------
// Elementwise f32 -> f16 conversion, 8 elements/thread (n multiple of 8).
// ---------------------------------------------------------------------------
__global__ void cvt_f32_f16(const float* __restrict__ x,
                            _Float16* __restrict__ y, int n) {
  const int i = (blockIdx.x * blockDim.x + threadIdx.x) * 8;
  if (i >= n) return;
  float4 a = *(const float4*)(x + i);
  float4 b = *(const float4*)(x + i + 4);
  v8h o;
  o[0] = (_Float16)a.x; o[1] = (_Float16)a.y; o[2] = (_Float16)a.z; o[3] = (_Float16)a.w;
  o[4] = (_Float16)b.x; o[5] = (_Float16)b.y; o[6] = (_Float16)b.z; o[7] = (_Float16)b.w;
  *(v8h*)(y + i) = o;
}

// ---------------------------------------------------------------------------
// Row softmax (4096 wide): one wave32 per row, 8 rows per block; probs in f16.
// ---------------------------------------------------------------------------
__global__ void softmax_kernel(const float* __restrict__ Sf,
                               _Float16* __restrict__ Ph) {
  const int wave = threadIdx.x >> 5;
  const int lane = threadIdx.x & 31;
  const int row = blockIdx.x * 8 + wave;
  const float* src = Sf + (size_t)row * SEQ;

  float m = -3.0e38f;
  for (int i = lane; i < SEQ; i += 32) m = fmaxf(m, src[i]);
#pragma unroll
  for (int off = 16; off > 0; off >>= 1) m = fmaxf(m, __shfl_xor(m, off, 32));

  float s = 0.0f;
  for (int i = lane; i < SEQ; i += 32) s += __expf(src[i] - m);
#pragma unroll
  for (int off = 16; off > 0; off >>= 1) s += __shfl_xor(s, off, 32);

  const float rinv = 1.0f / s;
  _Float16* dst = Ph + (size_t)row * SEQ;
  for (int i = lane; i < SEQ; i += 32)
    dst[i] = (_Float16)(__expf(src[i] - m) * rinv);
}

// ---------------------------------------------------------------------------
// Launcher.  Workspace layout (bytes), 192MB total:
//   [0,   32M)  Xh  f16 [B*S][D]      \ also: Sf f32 [S][S] = [0,64M) reused
//   [32M, 34M)  Wqh f16 [D][D]        |  per-batch AFTER the QKV stage
//   [34M, 36M)  Wkh f16 [D][D]        |  (time-disjoint on the stream)
//   [36M, 38M)  Wvh f16 [D][D]        /
//   [64M, 96M)  Qh  f16 [B*S][D]  (pre-scaled by 1/sqrt(D))
//   [96M,128M)  Kh  f16 [B*S][D]
//   [128M,160M) Vt  f16 [B][D][S]
//   [160M,192M) Ph  f16 [S][S]    (per-batch scratch)
// ---------------------------------------------------------------------------
extern "C" void kernel_launch(void* const* d_in, const int* in_sizes, int n_in,
                              void* d_out, int out_size, void* d_ws, size_t ws_size,
                              hipStream_t stream) {
  const float* X  = (const float*)d_in[0];
  const float* Wq = (const float*)d_in[1];
  const float* Wk = (const float*)d_in[2];
  const float* Wv = (const float*)d_in[3];
  float* out = (float*)d_out;

  char* ws = (char*)d_ws;
  _Float16* Xh  = (_Float16*)(ws);
  _Float16* Wqh = (_Float16*)(ws + ((size_t)32 << 20));
  _Float16* Wkh = (_Float16*)(ws + ((size_t)34 << 20));
  _Float16* Wvh = (_Float16*)(ws + ((size_t)36 << 20));
  float*    Sf  = (float*)   (ws);                       // reused after QKV
  _Float16* Qh  = (_Float16*)(ws + ((size_t)64 << 20));
  _Float16* Kh  = (_Float16*)(ws + ((size_t)96 << 20));
  _Float16* Vt  = (_Float16*)(ws + ((size_t)128 << 20));
  _Float16* Ph  = (_Float16*)(ws + ((size_t)160 << 20));

  const dim3 blk(256);
  const int nX = BATCH * SEQ * D_MODEL;   // 16,777,216
  const int nW = D_MODEL * D_MODEL;       // 1,048,576

  // f32 -> f16 pre-conversion (all GEMMs then share one tuned pipeline).
  cvt_f32_f16<<<dim3(nX / (8 * 256)), blk, 0, stream>>>(X, Xh, nX);
  cvt_f32_f16<<<dim3(nW / (8 * 256)), blk, 0, stream>>>(Wq, Wqh, nW);
  cvt_f32_f16<<<dim3(nW / (8 * 256)), blk, 0, stream>>>(Wk, Wkh, nW);
  cvt_f32_f16<<<dim3(nW / (8 * 256)), blk, 0, stream>>>(Wv, Wvh, nW);

  // QKV projections: M=16384, N=1024, K=1024.
  gemm_nt<1><<<dim3(128, 8), blk, 0, stream>>>(Xh, Wqh, Qh, D_MODEL, D_MODEL);
  gemm_nt<2><<<dim3(128, 8), blk, 0, stream>>>(Xh, Wkh, Kh, D_MODEL, D_MODEL);
  gemm_nt<3><<<dim3(128, 8), blk, 0, stream>>>(Xh, Wvh, Vt, D_MODEL, D_MODEL);

  for (int b = 0; b < BATCH; ++b) {
    const _Float16* Qb  = Qh + (size_t)b * SEQ * D_MODEL;
    const _Float16* Kb  = Kh + (size_t)b * SEQ * D_MODEL;
    const _Float16* Vtb = Vt + (size_t)b * D_MODEL * SEQ;
    float* Ob = out + (size_t)b * SEQ * D_MODEL;

    // scores: M=4096, N=4096, K=1024 -> Sf[q][k] (f32)
    gemm_nt<0><<<dim3(32, 32), blk, 0, stream>>>(Qb, Kb, Sf, D_MODEL, SEQ);
    // softmax rows -> probs f16
    softmax_kernel<<<dim3(512), blk, 0, stream>>>(Sf, Ph);
    // out: M=4096 (q), N=1024 (d), K=4096 (keys) -> Ob[q][d] (f32)
    gemm_nt<0><<<dim3(32, 8), blk, 0, stream>>>(Ph, Vtb, Ob, SEQ, D_MODEL);
  }
}
